// OneHeadAttention_1666447311338
// MI455X (gfx1250) — matrix operand order repaired and checked by reading, not compiled, not run on hardware
//
#include <hip/hip_runtime.h>
#include <hip/hip_bf16.h>

#define B_ 512
#define T_ 256
#define C_ 384
#define V_ 65

typedef __attribute__((ext_vector_type(16))) _Float16 v16h;
typedef __attribute__((ext_vector_type(8)))  _Float16 v8h;
typedef __attribute__((ext_vector_type(8)))  float    v8f;

// ---- CDNA5 WMMA fragment helpers (wave32, 16x16x32 f16 -> f32) ------------
// A-matrix 16x32: lane holds row m=lane&15; halves 0..15 = K(koff..koff+15),
// koff = (lane<16)?0:16 -> two 16B LDS reads.
__device__ __forceinline__ v16h ldA(const _Float16* p) {
    union { v16h v; v8h h[2]; } u;
    u.h[0] = *(const v8h*)(p);
    u.h[1] = *(const v8h*)(p + 8);
    return u.v;
}
// B-matrix 32x16: lane holds col n=lane&15; halves 0..15 = K(koff..koff+15),
// koff = (lane<16)?0:16 -> two 16B LDS reads (row-major [n][k] storage).
__device__ __forceinline__ v16h ldB(const _Float16* p) {
    union { v16h v; v8h h[2]; } u;
    u.h[0] = *(const v8h*)(p);
    u.h[1] = *(const v8h*)(p + 8);
    return u.v;
}
__device__ __forceinline__ v8f wmma16(v16h a, v16h b, v8f c) {
    return __builtin_amdgcn_wmma_f32_16x16x32_f16(
        false, a, false, b, (short)0, c, false, false);
}

// ============================================================================
// Kernel 1: x = tok_emb[idx] + pos_emb (fused); q,k,v = x @ {Wq,Wk,Wv} (f16 out)
// Grid: 1024 blocks of 256 threads; each block: 128 rows x full 384 cols x 3 mats.
// LDS (dynamic): xs[128][384] f16 (96KB) + wt[128][384] f16 (96KB) = 192KB.
// ============================================================================
__global__ __launch_bounds__(256) void k_qkv(
    const int* __restrict__ idx, const float* __restrict__ tok,
    const float* __restrict__ pos, const float* __restrict__ Wq,
    const float* __restrict__ Wk, const float* __restrict__ Wv,
    _Float16* __restrict__ qo, _Float16* __restrict__ ko,
    _Float16* __restrict__ vo)
{
    extern __shared__ char smem[];
    _Float16* xs = (_Float16*)smem;                       // [128][384]
    _Float16* wt = (_Float16*)(smem + 128 * C_ * 2);      // [128][384], wt[n][k]=W[k][colBase+n]

    const int tid = threadIdx.x;
    const int rowBase = blockIdx.x * 128;

    // Fused embedding gather into LDS x-tile (f16)
    for (int i = tid; i < 128 * C_; i += 256) {
        int r = i / C_, c = i - r * C_;
        int g = rowBase + r;
        int b = g >> 8, t = g & 255;
        int tk = idx[(b << 8) + t];
        xs[i] = (_Float16)(tok[tk * C_ + c] + pos[t * C_ + c]);
    }

    const int wave = tid >> 5, lane = tid & 31;
    const int wr = (wave >> 1) << 5;          // 0,32,64,96
    const int wc = (wave & 1) << 6;           // 0,64
    const int am = lane & 15;
    const int akoff = (lane < 16) ? 0 : 16;
    const int bkoff = (lane < 16) ? 0 : 16;
    const int mo = (lane < 16) ? 0 : 8;

    for (int m = 0; m < 3; ++m) {
        const float* W = (m == 0) ? Wq : (m == 1 ? Wk : Wv);
        _Float16* O   = (m == 0) ? qo : (m == 1 ? ko : vo);
        const float scl = (m == 0) ? 0.05103103630798287f : 1.0f; // fold C^-0.5 into q
        for (int cc = 0; cc < 3; ++cc) {
            const int colBase = cc * 128;
            __syncthreads();   // wt reuse / xs ready
            for (int i = tid; i < 128 * C_; i += 256) {
                int n = i / C_, k = i - n * C_;
                wt[i] = (_Float16)W[k * C_ + colBase + n];
            }
            __syncthreads();

            v8f acc[2][4];
            #pragma unroll
            for (int r2 = 0; r2 < 2; ++r2)
                #pragma unroll
                for (int t4 = 0; t4 < 4; ++t4) acc[r2][t4] = (v8f){};

            for (int k = 0; k < C_; k += 32) {
                v16h a0 = ldA(xs + (wr + am) * C_ + k + akoff);
                v16h a1 = ldA(xs + (wr + 16 + am) * C_ + k + akoff);
                #pragma unroll
                for (int t4 = 0; t4 < 4; ++t4) {
                    v16h bf = ldB(wt + (wc + t4 * 16 + am) * C_ + k + bkoff);
                    acc[0][t4] = wmma16(a0, bf, acc[0][t4]);
                    acc[1][t4] = wmma16(a1, bf, acc[1][t4]);
                }
            }
            #pragma unroll
            for (int r2 = 0; r2 < 2; ++r2)
                #pragma unroll
                for (int t4 = 0; t4 < 4; ++t4) {
                    int col = colBase + wc + t4 * 16 + am;
                    #pragma unroll
                    for (int j = 0; j < 8; ++j) {
                        int row = rowBase + wr + r2 * 16 + mo + j;
                        O[(size_t)row * C_ + col] = (_Float16)(acc[r2][t4][j] * scl);
                    }
                }
        }
    }
}

// ============================================================================
// Kernel 2: causal attention per (batch, 64-row q tile).
// Phase 1: S = q k^T via WMMA -> LDS f32; Phase 2: row softmax -> f16 P;
// Phase 3: O = P v via WMMA (v transposed into LDS, reusing S region).
// LDS: qs 48K | ks 48K | ps 32K | sS/vt 64K = 192KB (dynamic).
// ============================================================================
__global__ __launch_bounds__(256) void k_attn(
    const _Float16* __restrict__ q, const _Float16* __restrict__ k,
    const _Float16* __restrict__ v, _Float16* __restrict__ o)
{
    extern __shared__ char smem[];
    _Float16* qs = (_Float16*)smem;                 // [64][384]
    _Float16* ks = (_Float16*)(smem + 49152);       // [64][384]
    _Float16* ps = (_Float16*)(smem + 98304);       // [64][256]
    float*    sS = (float*)(smem + 131072);         // [64][256] f32
    _Float16* vt = (_Float16*)(smem + 131072);      // [128][256] f16 (aliases sS)

    const int b = blockIdx.y;
    const int qt = blockIdx.x;            // 0..3
    const int qBase = qt * 64;
    const int kmax = (qt + 1) * 64;       // causal: keys beyond diagonal tile skipped
    const int tid = threadIdx.x;
    const int wave = tid >> 5, lane = tid & 31;
    const int am = lane & 15;
    const int akoff = (lane < 16) ? 0 : 16;
    const int bkoff = (lane < 16) ? 0 : 16;
    const int mo = (lane < 16) ? 0 : 8;
    const int rg = (wave >> 1) << 4;      // this wave's 16-row group (0..48)

    // load q tile
    const _Float16* qg = q + ((size_t)b * T_ + qBase) * C_;
    for (int i = tid; i < 64 * C_; i += 256) qs[i] = qg[i];
    __syncthreads();

    // cache all 12 A-fragments of this wave's q rows (K=384)
    v16h afr[12];
    #pragma unroll
    for (int kk = 0; kk < 12; ++kk)
        afr[kk] = ldA(qs + (rg + am) * C_ + kk * 32 + akoff);

    // ---- Phase 1: scores ----
    const int cg = (wave & 1) * 32;
    for (int kt = 0; kt <= qt; ++kt) {
        __syncthreads();  // ks reuse
        const _Float16* kg = k + ((size_t)b * T_ + kt * 64) * C_;
        for (int i = tid; i < 64 * C_; i += 256) ks[i] = kg[i];
        __syncthreads();

        v8f acc0 = (v8f){}, acc1 = (v8f){};
        #pragma unroll
        for (int kk = 0; kk < 12; ++kk) {
            v16h b0 = ldB(ks + (cg + am) * C_ + kk * 32 + bkoff);
            v16h b1 = ldB(ks + (cg + 16 + am) * C_ + kk * 32 + bkoff);
            acc0 = wmma16(afr[kk], b0, acc0);
            acc1 = wmma16(afr[kk], b1, acc1);
        }
        #pragma unroll
        for (int j = 0; j < 8; ++j) {
            sS[(rg + mo + j) * 256 + kt * 64 + cg + am]      = acc0[j];
            sS[(rg + mo + j) * 256 + kt * 64 + cg + 16 + am] = acc1[j];
        }
    }
    __syncthreads();

    // ---- Phase 2: row softmax (wave handles 8 rows, lane-parallel cols) ----
    for (int r = wave * 8; r < wave * 8 + 8; ++r) {
        const int grow = qBase + r;
        float mx = -3.4e38f;
        for (int c = lane; c < kmax; c += 32) {
            float s = (c <= grow) ? sS[r * 256 + c] : -3.4e38f;
            mx = fmaxf(mx, s);
        }
        #pragma unroll
        for (int off = 16; off > 0; off >>= 1) mx = fmaxf(mx, __shfl_xor(mx, off, 32));
        float sum = 0.f;
        for (int c = lane; c < kmax; c += 32) {
            float p = (c <= grow) ? __expf(sS[r * 256 + c] - mx) : 0.f;
            sS[r * 256 + c] = p;
            sum += p;
        }
        #pragma unroll
        for (int off = 16; off > 0; off >>= 1) sum += __shfl_xor(sum, off, 32);
        const float inv = 1.f / sum;
        for (int c = lane; c < kmax; c += 32)
            ps[r * 256 + c] = (_Float16)(sS[r * 256 + c] * inv);
    }
    __syncthreads();

    // cache P A-fragments for this wave's rows (variable K, fully unrolled)
    const int nks = kmax >> 5;   // 2,4,6,8 K-steps of 32
    v16h pfr[8];
    #pragma unroll
    for (int kk = 0; kk < 8; ++kk)
        if (kk < nks) pfr[kk] = ldA(ps + (rg + am) * 256 + kk * 32 + akoff);

    // ---- Phase 3: O = P @ v, feature chunks of 128 ----
    const int ch = (wave & 1) * 64;
    for (int fc = 0; fc < 3; ++fc) {
        __syncthreads();  // vt reuse (first pass: sS dead)
        for (int i = tid; i < 128 * 256; i += 256) {
            int f = i >> 8, key = i & 255;
            if (key < kmax)
                vt[f * 256 + key] = v[((size_t)b * T_ + key) * C_ + fc * 128 + f];
        }
        __syncthreads();

        v8f oc[4];
        #pragma unroll
        for (int t4 = 0; t4 < 4; ++t4) oc[t4] = (v8f){};
        #pragma unroll
        for (int kk = 0; kk < 8; ++kk) {
            if (kk < nks) {
                #pragma unroll
                for (int t4 = 0; t4 < 4; ++t4) {
                    v16h bf = ldB(vt + (ch + t4 * 16 + am) * 256 + kk * 32 + bkoff);
                    oc[t4] = wmma16(pfr[kk], bf, oc[t4]);
                }
            }
        }
        #pragma unroll
        for (int t4 = 0; t4 < 4; ++t4)
            #pragma unroll
            for (int j = 0; j < 8; ++j) {
                int row = qBase + rg + mo + j;
                o[((size_t)b * T_ + row) * C_ + fc * 128 + ch + t4 * 16 + am] =
                    (_Float16)oc[t4][j];
            }
    }
}

// ============================================================================
// Kernel 3: logits = out @ Wo + bo ([131072,384]x[384,65], N padded to 80).
// LDS: os[128][384] f16 (96KB) + wot[80][384] f16 (60KB) = 156KB (dynamic).
// ============================================================================
__global__ __launch_bounds__(256) void k_logits(
    const _Float16* __restrict__ o, const float* __restrict__ Wo,
    const float* __restrict__ bo, float* __restrict__ out)
{
    extern __shared__ char smem[];
    _Float16* os  = (_Float16*)smem;             // [128][384]
    _Float16* wot = (_Float16*)(smem + 98304);   // [80][384], wot[n][k]=Wo[k][n]

    const int tid = threadIdx.x;
    const int rowBase = blockIdx.x * 128;

    for (int i = tid; i < 128 * C_; i += 256) os[i] = o[(size_t)rowBase * C_ + i];
    for (int i = tid; i < 80 * C_; i += 256) {
        int n = i / C_, k = i - n * C_;
        wot[i] = (n < V_) ? (_Float16)Wo[k * V_ + n] : (_Float16)0.f;
    }
    __syncthreads();

    const int wave = tid >> 5, lane = tid & 31;
    const int rg = wave << 4;                 // 16 rows per wave
    const int am = lane & 15;
    const int akoff = (lane < 16) ? 0 : 16;
    const int bkoff = (lane < 16) ? 0 : 16;
    const int mo = (lane < 16) ? 0 : 8;

    v16h afr[12];
    #pragma unroll
    for (int kk = 0; kk < 12; ++kk)
        afr[kk] = ldA(os + (rg + am) * C_ + kk * 32 + akoff);

    for (int ct = 0; ct < 5; ++ct) {
        v8f acc = (v8f){};
        #pragma unroll
        for (int kk = 0; kk < 12; ++kk) {
            v16h bf = ldB(wot + (ct * 16 + am) * C_ + kk * 32 + bkoff);
            acc = wmma16(afr[kk], bf, acc);
        }
        const int col = ct * 16 + am;
        if (col < V_) {
            const float bias = bo[col];
            #pragma unroll
            for (int j = 0; j < 8; ++j) {
                int row = rowBase + rg + mo + j;
                out[(size_t)row * V_ + col] = acc[j] + bias;
            }
        }
    }
}

// ============================================================================
extern "C" void kernel_launch(void* const* d_in, const int* in_sizes, int n_in,
                              void* d_out, int out_size, void* d_ws, size_t ws_size,
                              hipStream_t stream)
{
    const int*   idx = (const int*)d_in[0];
    const float* tok = (const float*)d_in[1];
    const float* pos = (const float*)d_in[2];
    const float* Wq  = (const float*)d_in[3];
    const float* Wk  = (const float*)d_in[4];
    const float* Wv  = (const float*)d_in[5];
    const float* Wo  = (const float*)d_in[6];
    const float* bo  = (const float*)d_in[7];
    float* out = (float*)d_out;

    const size_t elems = (size_t)B_ * T_ * C_;   // 50,331,648
    _Float16* qb = (_Float16*)d_ws;
    _Float16* kb = qb + elems;
    _Float16* vb = kb + elems;
    _Float16* ob = vb + elems;                   // total ws: 4*elems*2B = 384MiB

    k_qkv   <<<dim3((B_ * T_) / 128), 256, 196608, stream>>>(idx, tok, pos, Wq, Wk, Wv, qb, kb, vb);
    k_attn  <<<dim3(4, B_),           256, 196608, stream>>>(qb, kb, vb, ob);
    k_logits<<<dim3((B_ * T_) / 128), 256, 159744, stream>>>(ob, Wo, bo, out);
}
